// AutoregressiveDecoder_37572373906223
// MI455X (gfx1250) — matrix operand order repaired and checked by reading, not compiled, not run on hardware
//
#include <hip/hip_runtime.h>
#include <hip/hip_bf16.h>
#include <math.h>

typedef __attribute__((ext_vector_type(16))) _Float16 v16h;
typedef __attribute__((ext_vector_type(8)))  _Float16 v8h;
typedef __attribute__((ext_vector_type(8)))  float    v8f;

#define BROWS 16384
#define DD    512
#define HH    512

// ALL_PERMS flattened
__constant__ int c_perms[18] = {0,1,2, 0,2,1, 1,0,2, 1,2,0, 2,0,1, 2,1,0};

__device__ __forceinline__ float gelu_exact(float x) {
    return 0.5f * x * (1.0f + erff(x * 0.70710678118654752f));
}

// ---------------------------------------------------------------------------
// f32 -> f16 convert
// ---------------------------------------------------------------------------
__global__ void __launch_bounds__(256) k_convert_f16(const float* __restrict__ src,
                                                     _Float16* __restrict__ dst, int n) {
    int i = blockIdx.x * blockDim.x + threadIdx.x;
    int stride = gridDim.x * blockDim.x;
    for (; i < n; i += stride) dst[i] = (_Float16)src[i];
}

// ---------------------------------------------------------------------------
// WMMA GEMM: C[M x N] = act(A[M x K] @ W[K x N] + bias)   (f16 in, f32 acc/out)
//
// Block tile 64(M) x 128(N), 256 threads = 8 waves.
// Wave w owns a 16 x 64 strip: tile_m = (w&3)*16, tile_n = (w>>2)*64.
// Per 32-wide K-chunk each wave loads ONE A fragment (reused) + four B
// fragments and issues FOUR v_wmma_f32_16x16x32_f16.
// Double-buffered LDS: stage chunk k+1 while computing chunk k -> one
// workgroup barrier per K-chunk; global loads overlap WMMA via split
// LOADcnt/DScnt counters.
// LDS: A row-major (64x32 halfs), B transposed (128 n-rows x 32 k) x2 = 24KB.
// Fragments use contiguous k order (k = 16*(lane/16) + e) (16-bit A 16x32; B mirrored).
// ---------------------------------------------------------------------------
__global__ void __launch_bounds__(256) k_gemm_wmma_f16(
    const _Float16* __restrict__ A, const _Float16* __restrict__ W,
    const float* __restrict__ bias, float* __restrict__ C,
    int M, int N, int K, int do_gelu)
{
    __shared__ __align__(16) _Float16 sA[2][64 * 32];
    __shared__ __align__(16) _Float16 sBT[2][128 * 32];

    const int tid  = threadIdx.x;
    const int wid  = tid >> 5;
    const int lane = tid & 31;
    const int m0 = blockIdx.x * 64;
    const int n0 = blockIdx.y * 128;
    const int tm = (wid & 3) * 16;    // 0..48
    const int tn = (wid >> 2) * 64;   // 0 or 64

    // per-lane fragment addressing (contiguous k order, 16-bit operands)
    const int mA  = lane & 15;
    const int kbA = (lane < 16) ? 0 : 16;   // A: half-wave K halves
    const int nB  = lane & 15;
    const int kbB = (lane < 16) ? 0 : 16;   // B: half-wave K halves

    // staging indices
    const int ra = (tid * 8) >> 5;          // A row 0..63
    const int ka = (tid * 8) & 31;          // A k multiple of 8
    const int kb = (tid * 16) >> 7;         // B k 0..31
    const int nb = (tid * 16) & 127;        // B n multiple of 16

    v8f acc0 = {}, acc1 = {}, acc2 = {}, acc3 = {};

    auto stage = [&](int kc, int buf) {
        // A tile: 64x32 halfs, 8 per thread (one b128 load, contiguous store)
        *(v8h*)(sA[buf] + ra * 32 + ka) =
            *(const v8h*)(A + (size_t)(m0 + ra) * K + (kc + ka));
        // B tile transposed: 32(k) x 128(n), 16 per thread
        v8h w0 = *(const v8h*)(W + (size_t)(kc + kb) * N + (n0 + nb));
        v8h w1 = *(const v8h*)(W + (size_t)(kc + kb) * N + (n0 + nb + 8));
#pragma unroll
        for (int i = 0; i < 8; ++i) {
            sBT[buf][(nb + i) * 32 + kb]     = w0[i];
            sBT[buf][(nb + 8 + i) * 32 + kb] = w1[i];
        }
    };

    stage(0, 0);
    __syncthreads();

    int buf = 0;
    for (int kc = 0; kc < K; kc += 32) {
        if (kc + 32 < K) {
            stage(kc + 32, buf ^ 1);   // prefetch next chunk into other buffer
            __builtin_prefetch((const void*)(A + (size_t)(m0 + ra) * K + kc + 64), 0, 1);
        }

        union { v16h v; v8h h[2]; } af, bf0, bf1, bf2, bf3;
        // A frag: e -> k = kbA + e  (two b128 ds loads)
        const _Float16* ap = sA[buf] + (tm + mA) * 32;
        af.h[0] = *(const v8h*)(ap + kbA);
        af.h[1] = *(const v8h*)(ap + kbA + 8);
        // B frags (transposed tile): e -> k = kbB + e
        const _Float16* bp = sBT[buf] + (tn + nB) * 32 + kbB;
        bf0.h[0] = *(const v8h*)(bp);
        bf0.h[1] = *(const v8h*)(bp + 8);
        bf1.h[0] = *(const v8h*)(bp + 16 * 32);
        bf1.h[1] = *(const v8h*)(bp + 16 * 32 + 8);
        bf2.h[0] = *(const v8h*)(bp + 32 * 32);
        bf2.h[1] = *(const v8h*)(bp + 32 * 32 + 8);
        bf3.h[0] = *(const v8h*)(bp + 48 * 32);
        bf3.h[1] = *(const v8h*)(bp + 48 * 32 + 8);

        acc0 = __builtin_amdgcn_wmma_f32_16x16x32_f16(false, af.v, false, bf0.v,
                                                      (short)0, acc0, false, false);
        acc1 = __builtin_amdgcn_wmma_f32_16x16x32_f16(false, af.v, false, bf1.v,
                                                      (short)0, acc1, false, false);
        acc2 = __builtin_amdgcn_wmma_f32_16x16x32_f16(false, af.v, false, bf2.v,
                                                      (short)0, acc2, false, false);
        acc3 = __builtin_amdgcn_wmma_f32_16x16x32_f16(false, af.v, false, bf3.v,
                                                      (short)0, acc3, false, false);

        __syncthreads();   // next-buffer writes + this-buffer reads both done
        buf ^= 1;
    }

    // ---- epilogue: C/D layout (VGPR r = M row; lanes 16-31 -> M+8) ----
    const int gm_base = m0 + tm + ((lane < 16) ? 0 : 8);
    v8f accs[4] = {acc0, acc1, acc2, acc3};
#pragma unroll
    for (int j = 0; j < 4; ++j) {
        const int gn = n0 + tn + j * 16 + (lane & 15);
        const float bv = bias[gn];
#pragma unroll
        for (int r = 0; r < 8; ++r) {
            float v = accs[j][r] + bv;
            if (do_gelu) v = gelu_exact(v);
            C[(size_t)(gm_base + r) * N + gn] = v;
        }
    }
}

// ---------------------------------------------------------------------------
// Per-step layer-1: h1 = gelu(base + state @ w1[D:D+s] + onehot row)  -> f16
// ---------------------------------------------------------------------------
__global__ void __launch_bounds__(512) k_step_h1(
    const float* __restrict__ base_p, const float* __restrict__ base_f,
    const float* __restrict__ pw1, const float* __restrict__ fw1,
    const float* __restrict__ pres_state, const float* __restrict__ full_state,
    const int* __restrict__ perm_idx, int step,
    _Float16* __restrict__ h1p, _Float16* __restrict__ h1f)
{
    const int row = blockIdx.x;
    const int col = threadIdx.x;
    const int ridx = c_perms[perm_idx[row] * 3 + step];

    const float* ps = pres_state + (size_t)row * 6;
    const float* fs = full_state + (size_t)row * 12;

    float cp = pw1[(size_t)(DD + 6 + ridx) * HH + col];   // one-hot row
#pragma unroll
    for (int j = 0; j < 6; ++j)
        cp += ps[j] * pw1[(size_t)(DD + j) * HH + col];

    float cf = fw1[(size_t)(DD + 12 + ridx) * HH + col];
#pragma unroll
    for (int j = 0; j < 12; ++j)
        cf += fs[j] * fw1[(size_t)(DD + j) * HH + col];

    const size_t o = (size_t)row * HH + col;
    h1p[o] = (_Float16)gelu_exact(base_p[o] + cp);
    h1f[o] = (_Float16)gelu_exact(base_f[o] + cf);
}

// ---------------------------------------------------------------------------
// Per-step head: layer-3 dots, losses, state + output updates. Wave per row.
// ---------------------------------------------------------------------------
__global__ void __launch_bounds__(256) k_step_head(
    const float* __restrict__ h2p, const float* __restrict__ h2f,
    const float* __restrict__ pw3, const float* __restrict__ pb3,
    const float* __restrict__ fw3, const float* __restrict__ fb3,
    const float* __restrict__ freq, const float* __restrict__ pres,
    const float* __restrict__ enrich,
    const int* __restrict__ round_mask, const int* __restrict__ perm_idx, int step,
    float* __restrict__ pres_state, float* __restrict__ full_state,
    float* __restrict__ losspart, float* __restrict__ dout)
{
    const int lane = threadIdx.x & 31;
    const int row  = blockIdx.x * 8 + (threadIdx.x >> 5);

    float sp = 0.f, sf = 0.f, se = 0.f;
    for (int i = lane; i < HH / 2; i += 32)
        sp += h2p[(size_t)row * (HH / 2) + i] * pw3[i];
    for (int i = lane; i < HH; i += 32) {
        float h = h2f[(size_t)row * HH + i];
        sf += h * fw3[2 * i];
        se += h * fw3[2 * i + 1];
    }
#pragma unroll
    for (int off = 16; off > 0; off >>= 1) {
        sp += __shfl_xor(sp, off, 32);
        sf += __shfl_xor(sf, off, 32);
        se += __shfl_xor(se, off, 32);
    }

    if (lane == 0) {
        const int ridx = c_perms[perm_idx[row] * 3 + step];
        const float pp = sp + pb3[0];
        const float pf = sf + fb3[0];
        const float pe = se + fb3[1];

        const float m    = (float)round_mask[row * 3 + ridx];
        const float gt_f = freq[row * 3 + ridx];
        const float gt_p = pres[row * 3 + ridx];
        const float gt_e = enrich[row * 3 + ridx];

        const float bce = fmaxf(pp, 0.f) - pp * gt_p + log1pf(expf(-fabsf(pp)));
        losspart[(size_t)row * 4 + 0] = (pf - gt_f) * (pf - gt_f) * m;
        losspart[(size_t)row * 4 + 1] = bce * m;
        losspart[(size_t)row * 4 + 2] = (pe - gt_e) * (pe - gt_e) * m;
        losspart[(size_t)row * 4 + 3] = m;

        const bool msk = m > 0.5f;
        const float act_f = msk ? fminf(fmaxf(pf, -10.f), 10.f) : gt_f;
        const float act_p = msk ? (1.f / (1.f + expf(-pp))) : gt_p;
        const float act_e = msk ? fminf(fmaxf(pe, -100.f), 100.f) : gt_e;

        pres_state[(size_t)row * 6 + ridx * 2 + 0] = act_p;
        pres_state[(size_t)row * 6 + ridx * 2 + 1] = 1.f;
        full_state[(size_t)row * 12 + ridx * 4 + 0] = act_f;
        full_state[(size_t)row * 12 + ridx * 4 + 1] = act_p;
        full_state[(size_t)row * 12 + ridx * 4 + 2] = act_e;
        full_state[(size_t)row * 12 + ridx * 4 + 3] = 1.f;

        // out layout: [3 losses][d_f 49152][d_p 49152][d_e 49152]
        dout[3 + 0 * (BROWS * 3) + row * 3 + ridx] = act_f;
        dout[3 + 1 * (BROWS * 3) + row * 3 + ridx] = act_p;
        dout[3 + 2 * (BROWS * 3) + row * 3 + ridx] = act_e;
    }
}

// ---------------------------------------------------------------------------
// Deterministic single-block tree reduction of per-row loss parts into acc[4]
// ---------------------------------------------------------------------------
__global__ void __launch_bounds__(1024) k_reduce_loss(const float* __restrict__ losspart,
                                                      float* __restrict__ acc)
{
    __shared__ float s0[1024], s1[1024], s2[1024], s3[1024];
    const int tid = threadIdx.x;
    float v0 = 0.f, v1 = 0.f, v2 = 0.f, v3 = 0.f;
    for (int r = tid; r < BROWS; r += 1024) {
        v0 += losspart[(size_t)r * 4 + 0];
        v1 += losspart[(size_t)r * 4 + 1];
        v2 += losspart[(size_t)r * 4 + 2];
        v3 += losspart[(size_t)r * 4 + 3];
    }
    s0[tid] = v0; s1[tid] = v1; s2[tid] = v2; s3[tid] = v3;
    __syncthreads();
    for (int off = 512; off > 0; off >>= 1) {
        if (tid < off) {
            s0[tid] += s0[tid + off];
            s1[tid] += s1[tid + off];
            s2[tid] += s2[tid + off];
            s3[tid] += s3[tid + off];
        }
        __syncthreads();
    }
    if (tid == 0) {
        acc[0] += s0[0]; acc[1] += s1[0]; acc[2] += s2[0]; acc[3] += s3[0];
    }
}

__global__ void k_finalize(const float* __restrict__ acc, float* __restrict__ dout) {
    if (threadIdx.x == 0 && blockIdx.x == 0) {
        const float nm = acc[3] + 1e-8f;
        dout[0] = acc[0] / nm;
        dout[1] = acc[1] / nm;
        dout[2] = acc[2] / nm;
    }
}

// ---------------------------------------------------------------------------
// Host launcher
// ---------------------------------------------------------------------------
extern "C" void kernel_launch(void* const* d_in, const int* in_sizes, int n_in,
                              void* d_out, int out_size, void* d_ws, size_t ws_size,
                              hipStream_t stream) {
    const float* seq    = (const float*)d_in[0];
    const float* freq   = (const float*)d_in[1];
    const float* pres   = (const float*)d_in[2];
    const float* enrich = (const float*)d_in[3];
    const float* pw1 = (const float*)d_in[4];
    const float* pb1 = (const float*)d_in[5];
    const float* pw2 = (const float*)d_in[6];
    const float* pb2 = (const float*)d_in[7];
    const float* pw3 = (const float*)d_in[8];
    const float* pb3 = (const float*)d_in[9];
    const float* fw1 = (const float*)d_in[10];
    const float* fb1 = (const float*)d_in[11];
    const float* fw2 = (const float*)d_in[12];
    const float* fb2 = (const float*)d_in[13];
    const float* fw3 = (const float*)d_in[14];
    const float* fb3 = (const float*)d_in[15];
    const int* perm_idx   = (const int*)d_in[16];
    const int* round_mask = (const int*)d_in[17];
    float* out = (float*)d_out;

    // ---- workspace layout (256B aligned regions) ----
    char* ws = (char*)d_ws;
    size_t off = 0;
    auto carve = [&](size_t bytes) -> char* {
        char* p = ws + off;
        off += (bytes + 255) & ~(size_t)255;
        return p;
    };
    _Float16* seqh  = (_Float16*)carve((size_t)BROWS * DD * 2);
    _Float16* pw1h  = (_Float16*)carve((size_t)DD * HH * 2);
    _Float16* fw1h  = (_Float16*)carve((size_t)DD * HH * 2);
    _Float16* pw2h  = (_Float16*)carve((size_t)HH * (HH / 2) * 2);
    _Float16* fw2h  = (_Float16*)carve((size_t)HH * HH * 2);
    float* base_p   = (float*)carve((size_t)BROWS * HH * 4);
    float* base_f   = (float*)carve((size_t)BROWS * HH * 4);
    _Float16* h1p   = (_Float16*)carve((size_t)BROWS * HH * 2);
    _Float16* h1f   = (_Float16*)carve((size_t)BROWS * HH * 2);
    float* h2p      = (float*)carve((size_t)BROWS * (HH / 2) * 4);
    float* h2f      = (float*)carve((size_t)BROWS * HH * 4);
    float* pres_st  = (float*)carve((size_t)BROWS * 6 * 4);
    float* full_st  = (float*)carve((size_t)BROWS * 12 * 4);
    float* losspart = (float*)carve((size_t)BROWS * 4 * 4);
    float* acc      = (float*)carve(4 * 4);
    (void)ws_size; (void)in_sizes; (void)n_in;

    // ---- init (graph-capturable async memsets) ----
    hipMemsetAsync(acc, 0, 4 * 4, stream);
    hipMemsetAsync(pres_st, 0, (size_t)BROWS * 6 * 4, stream);
    hipMemsetAsync(full_st, 0, (size_t)BROWS * 12 * 4, stream);
    hipMemsetAsync(d_out, 0, (size_t)out_size * 4, stream);

    // ---- f32 -> f16 conversions ----
    auto conv = [&](const float* s, _Float16* d, int n) {
        int blocks = (n + 256 * 8 - 1) / (256 * 8);
        k_convert_f16<<<blocks, 256, 0, stream>>>(s, d, n);
    };
    conv(seq, seqh, BROWS * DD);
    conv(pw1, pw1h, DD * HH);          // first 512 rows of pw1 (contiguous)
    conv(fw1, fw1h, DD * HH);          // first 512 rows of fw1
    conv(pw2, pw2h, HH * (HH / 2));
    conv(fw2, fw2h, HH * HH);

    // ---- base GEMMs (shared across all 3 steps): no activation ----
    {
        dim3 grid(BROWS / 64, HH / 128);
        k_gemm_wmma_f16<<<grid, 256, 0, stream>>>(seqh, pw1h, pb1, base_p,
                                                  BROWS, HH, DD, 0);
        k_gemm_wmma_f16<<<grid, 256, 0, stream>>>(seqh, fw1h, fb1, base_f,
                                                  BROWS, HH, DD, 0);
    }

    // ---- 3 sequential decode steps ----
    for (int step = 0; step < 3; ++step) {
        k_step_h1<<<BROWS, 512, 0, stream>>>(base_p, base_f, pw1, fw1,
                                             pres_st, full_st, perm_idx, step,
                                             h1p, h1f);

        dim3 gp(BROWS / 64, (HH / 2) / 128);
        k_gemm_wmma_f16<<<gp, 256, 0, stream>>>(h1p, pw2h, pb2, h2p,
                                                BROWS, HH / 2, HH, 1);
        dim3 gf(BROWS / 64, HH / 128);
        k_gemm_wmma_f16<<<gf, 256, 0, stream>>>(h1f, fw2h, fb2, h2f,
                                                BROWS, HH, HH, 1);

        k_step_head<<<BROWS / 8, 256, 0, stream>>>(h2p, h2f, pw3, pb3, fw3, fb3,
                                                   freq, pres, enrich,
                                                   round_mask, perm_idx, step,
                                                   pres_st, full_st, losspart, out);

        k_reduce_loss<<<1, 1024, 0, stream>>>(losspart, acc);
    }

    k_finalize<<<1, 1, 0, stream>>>(acc, out);
}
